// RegressionBranch_76192719831673
// MI455X (gfx1250) — matrix operand order repaired and checked by reading, not compiled, not run on hardware
//
#include <hip/hip_runtime.h>

typedef __attribute__((ext_vector_type(16))) _Float16 v16h;
typedef __attribute__((ext_vector_type(8)))  float    v8f;

#define N_NODES 100000
#define N_EDGES 1600000
#define IN_FEAT 128

// ---------------------------------------------------------------------------
// Workspace layout (floats / halves):
//   hn_aggr : [N][128] f32                      (51.2 MB)
//   he_aggr : [N]      f32                      (0.4 MB)
//   bfrag   : 12 * 8 * 32 * 16 halves           (96 KB)  WMMA B fragments
//             tiles 0..3  : W1 rows   0..127 (hn part),   k-tiles of 32
//             tiles 4..7  : W1 rows 128..255 (aggr part)
//             tiles 8..11 : W2 rows   0..127
// ---------------------------------------------------------------------------

__global__ void zero_kernel(float* __restrict__ p, long long n) {
  long long i = (long long)blockIdx.x * blockDim.x + threadIdx.x;
  if (i < n) p[i] = 0.0f;
}

// Pack weights into WMMA B-fragment layout.
// B (32x16, f16): lane l holds column n = l&15, group g = l>>4;
// half h<8 holds K = g*8 + h, h>=8 holds K = 16 + g*8 + (h-8) of the 32-K tile.
__global__ void prep_frags(const float* __restrict__ W1,
                           const float* __restrict__ W2,
                           _Float16* __restrict__ frag) {
  int t = blockIdx.x * blockDim.x + threadIdx.x;   // 0 .. 3071
  if (t >= 12 * 8 * 32) return;
  int lane = t & 31;
  int nt   = (t >> 5) & 7;
  int tile = t >> 8;          // 0..11
  int part = tile >> 2;       // 0: W1-hn, 1: W1-aggr, 2: W2
  int kt   = tile & 3;
  int g    = lane >> 4;
  int n    = nt * 16 + (lane & 15);
#pragma unroll
  for (int h = 0; h < 16; ++h) {
    int kg = kt * 32 + (h < 8 ? g * 8 + h : 16 + g * 8 + (h - 8));  // K within 0..127
    float v;
    if (part == 0)      v = W1[kg * 128 + n];
    else if (part == 1) v = W1[(128 + kg) * 128 + n];
    else                v = W2[kg * 128 + n];
    frag[t * 16 + h] = (_Float16)v;
  }
}

// One wave (32 lanes) per edge: coalesced 512B gather + f32 global atomics.
__global__ void scatter_kernel(const float* __restrict__ hn,
                               const float* __restrict__ he,
                               const int*   __restrict__ src,
                               const int*   __restrict__ dst,
                               float* __restrict__ hn_aggr,
                               float* __restrict__ he_aggr) {
  int e = blockIdx.x * 8 + (threadIdx.x >> 5);
  if (e >= N_EDGES) return;
  int lane = threadIdx.x & 31;
  int s = src[e], d = dst[e];
  const float4 v = *(const float4*)(hn + (long long)s * IN_FEAT + lane * 4);
  float* p = hn_aggr + (long long)d * IN_FEAT + lane * 4;
  atomicAdd(p + 0, v.x);
  atomicAdd(p + 1, v.y);
  atomicAdd(p + 2, v.z);
  atomicAdd(p + 3, v.w);
  if (lane == 0) atomicAdd(he_aggr + d, he[e]);
}

// Fused 3-layer MLP, one wave per 16-node tile.
// A layout (16x32 f16): lane m = l&15, g = l>>4;
//   h<8  -> K = g*8 + h;  h>=8 -> K = 16 + g*8 + (h-8).
__global__ void __launch_bounds__(32) mlp_kernel(
    const float*     __restrict__ hn,
    const float*     __restrict__ hn_aggr,
    const float*     __restrict__ he_aggr,
    const _Float16*  __restrict__ bfrag,
    const float*     __restrict__ W1,   // for row 256 (he_aggr rank-1 term)
    const float*     __restrict__ b1,
    const float*     __restrict__ b2,
    const float*     __restrict__ W3,
    const float*     __restrict__ b3,
    float*           __restrict__ out) {
  __shared__ _Float16 lds_h[16 * 128];     // layer-1 activations (f16)
  __shared__ float    lds_s[16];           // he_aggr for this tile
  __shared__ float    lds_red[2][8][16];   // final reduction

  const int m0    = blockIdx.x * 16;
  const int lane  = threadIdx.x;
  const int g     = lane >> 4;
  const int idx16 = lane & 15;

  if (lane < 16) lds_s[lane] = he_aggr[m0 + lane];
  __syncthreads();

  // ---- build layer-1 A fragments (hn k-tiles 0..3, hn_aggr k-tiles 4..7) ----
  v16h a[8];
  const float* rowp_hn = hn      + (long long)(m0 + idx16) * IN_FEAT;
  const float* rowp_ag = hn_aggr + (long long)(m0 + idx16) * IN_FEAT;
#pragma unroll
  for (int kt = 0; kt < 4; ++kt) {
    const float* p0 = rowp_hn + kt * 32 + g * 8;        // h = 0..7
    const float* p1 = rowp_hn + kt * 32 + 16 + g * 8;   // h = 8..15
    v16h f;
#pragma unroll
    for (int h = 0; h < 8; ++h) { f[h] = (_Float16)p0[h]; f[h + 8] = (_Float16)p1[h]; }
    a[kt] = f;
    const float* q0 = rowp_ag + kt * 32 + g * 8;
    const float* q1 = rowp_ag + kt * 32 + 16 + g * 8;
    v16h f2;
#pragma unroll
    for (int h = 0; h < 8; ++h) { f2[h] = (_Float16)q0[h]; f2[h + 8] = (_Float16)q1[h]; }
    a[4 + kt] = f2;
  }

  const v16h* B = (const v16h*)bfrag;

  // ---- layer 1: [16x257] @ [257x128], +bias, ReLU -> LDS (f16) ----
  for (int nt = 0; nt < 8; ++nt) {
    v8f c = {};
#pragma unroll
    for (int kt = 0; kt < 4; ++kt)
      c = __builtin_amdgcn_wmma_f32_16x16x32_f16(
          false, a[kt], false, B[(kt * 8 + nt) * 32 + lane], (short)0, c, false, false);
#pragma unroll
    for (int kt = 0; kt < 4; ++kt)
      c = __builtin_amdgcn_wmma_f32_16x16x32_f16(
          false, a[4 + kt], false, B[((4 + kt) * 8 + nt) * 32 + lane], (short)0, c, false, false);
    int   n  = nt * 16 + idx16;
    float wl = W1[256 * 128 + n];     // row 256: he_aggr rank-1 term
    float bb = b1[n];
#pragma unroll
    for (int r = 0; r < 8; ++r) {
      int   M = r + g * 8;
      float v = c[r] + lds_s[M] * wl + bb;
      v = v > 0.0f ? v : 0.0f;
      lds_h[M * 128 + n] = (_Float16)v;
    }
  }
  __syncthreads();

  // ---- build layer-2 A fragments from LDS ----
  v16h a2[4];
  const _Float16* row2 = lds_h + idx16 * 128;
#pragma unroll
  for (int kt = 0; kt < 4; ++kt) {
    v16h f;
#pragma unroll
    for (int h = 0; h < 8; ++h) {
      f[h]     = row2[kt * 32 + g * 8 + h];
      f[h + 8] = row2[kt * 32 + 16 + g * 8 + h];
    }
    a2[kt] = f;
  }

  // ---- layer 2 (+bias, ReLU) fused with layer 3 (dot with W3) ----
  float acc[8];
#pragma unroll
  for (int r = 0; r < 8; ++r) acc[r] = 0.0f;

  for (int nt = 0; nt < 8; ++nt) {
    v8f c = {};
#pragma unroll
    for (int kt = 0; kt < 4; ++kt)
      c = __builtin_amdgcn_wmma_f32_16x16x32_f16(
          false, a2[kt], false, B[((8 + kt) * 8 + nt) * 32 + lane], (short)0, c, false, false);
    int   n  = nt * 16 + idx16;
    float bb = b2[n];
    float w3 = W3[n];
#pragma unroll
    for (int r = 0; r < 8; ++r) {
      float v = c[r] + bb;
      v = v > 0.0f ? v : 0.0f;
      acc[r] += v * w3;
    }
  }

#pragma unroll
  for (int r = 0; r < 8; ++r) lds_red[g][r][idx16] = acc[r];
  __syncthreads();

  if (lane < 16) {
    int M = lane;
    const float* rr = &lds_red[M >> 3][M & 7][0];
    float s = 0.0f;
#pragma unroll
    for (int j = 0; j < 16; ++j) s += rr[j];
    out[m0 + M] = s + b3[0];
  }
}

extern "C" void kernel_launch(void* const* d_in, const int* in_sizes, int n_in,
                              void* d_out, int out_size, void* d_ws, size_t ws_size,
                              hipStream_t stream) {
  const float* hn = (const float*)d_in[0];
  const float* he = (const float*)d_in[1];
  const int*   src = (const int*)d_in[2];
  const int*   dst = (const int*)d_in[3];
  const float* W1 = (const float*)d_in[4];
  const float* b1 = (const float*)d_in[5];
  const float* W2 = (const float*)d_in[6];
  const float* b2 = (const float*)d_in[7];
  const float* W3 = (const float*)d_in[8];
  const float* b3 = (const float*)d_in[9];
  float* out = (float*)d_out;

  float*    hn_aggr = (float*)d_ws;
  float*    he_aggr = hn_aggr + (size_t)N_NODES * IN_FEAT;
  _Float16* frag    = (_Float16*)(he_aggr + N_NODES);

  long long nz = (long long)N_NODES * IN_FEAT + N_NODES;
  zero_kernel<<<(int)((nz + 255) / 256), 256, 0, stream>>>(hn_aggr, nz);
  prep_frags<<<(12 * 8 * 32 + 255) / 256, 256, 0, stream>>>(W1, W2, frag);
  scatter_kernel<<<(N_EDGES + 7) / 8, 256, 0, stream>>>(hn, he, src, dst, hn_aggr, he_aggr);
  mlp_kernel<<<N_NODES / 16, 32, 0, stream>>>(hn, hn_aggr, he_aggr, frag,
                                              W1, b1, b2, W3, b3, out);
}
